// BayesianBVPMultiScaleGenerator_15152644621015
// MI455X (gfx1250) — matrix operand order repaired and checked by reading, not compiled, not run on hardware
//
#include <hip/hip_runtime.h>
#include <math.h>
#include <stdint.h>

#define H 256
#define TLEN 1024
#define BATCH 64
#define PI_F 3.14159265358979323846f

typedef _Float16 f16;
typedef __attribute__((ext_vector_type(16))) _Float16 v16h;
typedef __attribute__((ext_vector_type(8)))  _Float16 v8h;
typedef __attribute__((ext_vector_type(8)))  float    v8f;

// ---- gfx1250 async global->LDS path (guarded; falls back to VGPR staging) ----
#if defined(__has_builtin)
#if __has_builtin(__builtin_amdgcn_global_load_async_to_lds_b128) && \
    __has_builtin(__builtin_amdgcn_s_wait_asynccnt)
#define USE_ASYNC_LDS 1
#endif
#endif
#ifndef USE_ASYNC_LDS
#define USE_ASYNC_LDS 0
#endif

#if USE_ASYNC_LDS
// builtin signature: (v4i addrspace(1)*, v4i addrspace(3)*, imm offset, imm cpol)
typedef int v4i __attribute__((vector_size(16)));
typedef __attribute__((address_space(1))) v4i as1_v4i;
typedef __attribute__((address_space(3))) v4i as3_v4i;
__device__ __forceinline__ void async_ld_b128(const void* g, const void* l) {
    as1_v4i* gp = (as1_v4i*)(unsigned long long)(uintptr_t)g;
    // low 32 bits of a generic LDS address are the LDS byte offset
    as3_v4i* lp = (as3_v4i*)(unsigned)(uintptr_t)l;
    __builtin_amdgcn_global_load_async_to_lds_b128(gp, lp, 0, 0);
}
#endif

// ---------------------------------------------------------------- helpers
__device__ __forceinline__ v8f zero8() {
    v8f z;
#pragma unroll
    for (int i = 0; i < 8; ++i) z[i] = 0.f;
    return z;
}
__device__ __forceinline__ v16h cat8(v8h lo, v8h hi) {
    v16h r;
#pragma unroll
    for (int i = 0; i < 8; ++i) { r[i] = lo[i]; r[i + 8] = hi[i]; }
    return r;
}
// A fragment (16x32 f16) from row-major source (ld halves/row).
// Lane 0-15: M=lane, K {0..7, 16..23}; lane 16-31: M=lane-16, K {8..15, 24..31}.
__device__ __forceinline__ v16h frag_a(const f16* p, int ld, int m0, int k0) {
    int lane = threadIdx.x & 31;
    const f16* q = p + (size_t)(m0 + (lane & 15)) * ld + k0 + ((lane >> 4) << 3);
    v8h lo = *(const v8h*)q;
    v8h hi = *(const v8h*)(q + 16);
    return cat8(lo, hi);
}
// B fragment (32x16) from W stored row-major (N,K): B(k,n) = W[n][k].
// Lane 0-15: N=lane, K {0..7, 16..23}; lane 16-31: N=lane-16, K {8..15, 24..31}.
__device__ __forceinline__ v16h frag_b(const f16* p, int ld, int n0, int k0) {
    int lane = threadIdx.x & 31;
    const f16* q = p + (size_t)(n0 + (lane & 15)) * ld + k0 + ((lane >> 4) << 3);
    v8h lo = *(const v8h*)q;
    v8h hi = *(const v8h*)(q + 16);
    return cat8(lo, hi);
}
__device__ __forceinline__ v8f wmma16(v16h a, v16h b, v8f c) {
    return __builtin_amdgcn_wmma_f32_16x16x32_f16(false, a, false, b, (short)0, c, false, false);
}
__device__ __forceinline__ float sigm(float x)  { return 1.f / (1.f + __expf(-x)); }
__device__ __forceinline__ float lrelu(float x) { return x > 0.f ? x : 0.2f * x; }

// ---------------------------------------------------------------- packs
__global__ void k_f32_to_f16(const float* __restrict__ in, f16* __restrict__ out, int n) {
    int i = blockIdx.x * blockDim.x + threadIdx.x;
    if (i < n) out[i] = (f16)in[i];
}
__global__ void k_add2(const float* a, const float* b, float* o, int n) {
    int i = blockIdx.x * blockDim.x + threadIdx.x;
    if (i < n) o[i] = a[i] + b[i];
}
// conv weight (O,I,ks) -> wpk[o][k*I+i] f16 ; BN-eval + bias folded into scale/shift.
__global__ void k_pack_conv(const float* w, const float* cb, const float* g, const float* bb,
                            f16* wpk, float* scale, float* shift, int O, int I, int ks) {
    int n = O * I * ks;
    int idx = blockIdx.x * blockDim.x + threadIdx.x;
    if (idx < n) {
        int o = idx / (I * ks);
        int rem = idx % (I * ks);
        int k = rem / I;
        int i = rem % I;
        wpk[idx] = (f16)w[(size_t)(o * I + i) * ks + k];
    }
    if (idx < O) {
        float sc = g[idx] * rsqrtf(1.f + 1e-5f);
        scale[idx] = sc;
        shift[idx] = cb[idx] * sc + bb[idx];
    }
}

// ---------------------------------------------------------------- np head: h = lrelu(LN([z|le]@W.T+b)); sig16 = [h | le] f16
__global__ __launch_bounds__(256) void k_np_head(
    const float* z, const int* labels, const float* emb,
    const float* w, const float* bias, const float* g, const float* be, f16* sig16) {
    __shared__ float red[256];
    int b = blockIdx.x, o = threadIdx.x;
    int lab = labels[b];
    const float* wr = w + (size_t)o * 384;
    float acc = bias[o];
    for (int k = 0; k < 128; ++k) acc += z[b * 128 + k] * wr[k];
    for (int k = 0; k < 256; ++k) acc += emb[lab * 256 + k] * wr[128 + k];
    red[o] = acc; __syncthreads();
    for (int s = 128; s > 0; s >>= 1) { if (o < s) red[o] += red[o + s]; __syncthreads(); }
    float mean = red[0] * (1.f / 256.f); __syncthreads();
    float d = acc - mean;
    red[o] = d * d; __syncthreads();
    for (int s = 128; s > 0; s >>= 1) { if (o < s) red[o] += red[o + s]; __syncthreads(); }
    float var = red[0] * (1.f / 256.f);
    float hv = lrelu(d * rsqrtf(var + 1e-5f) * g[o] + be[o]);
    sig16[b * 512 + o]       = (f16)hv;
    sig16[b * 512 + 256 + o] = (f16)emb[lab * 256 + o];
}

// osc mid: lrelu(LN(h@osc_w1.T + b1)) -> f16
__global__ __launch_bounds__(256) void k_oscmid(
    const f16* sig16, const float* w1, const float* b1,
    const float* g, const float* be, f16* mid16) {
    __shared__ float red[256];
    int b = blockIdx.x, o = threadIdx.x;
    const float* wr = w1 + (size_t)o * 256;
    float acc = b1[o];
    for (int k = 0; k < 256; ++k) acc += (float)sig16[b * 512 + k] * wr[k];
    red[o] = acc; __syncthreads();
    for (int s = 128; s > 0; s >>= 1) { if (o < s) red[o] += red[o + s]; __syncthreads(); }
    float mean = red[0] * (1.f / 256.f); __syncthreads();
    float d = acc - mean;
    red[o] = d * d; __syncthreads();
    for (int s = 128; s > 0; s >>= 1) { if (o < s) red[o] += red[o + s]; __syncthreads(); }
    float var = red[0] * (1.f / 256.f);
    mid16[b * 256 + o] = (f16)lrelu(d * rsqrtf(var + 1e-5f) * g[o] + be[o]);
}

// ---------------------------------------------------------------- generic WMMA GEMM: C = act(A(MxK) @ W(N,K)^T + bias)
// A, W pre-packed f16. Block=256 thr (8 waves); block tile 64x64; each wave two 16x16 tiles.
__global__ __launch_bounds__(256) void k_gemm(
    const f16* __restrict__ A, const f16* __restrict__ W, const float* __restrict__ bias,
    float* outf, f16* outh, int N, int K, int act) {
    int n0b = blockIdx.x * 64, m0b = blockIdx.y * 64;
    int tid = threadIdx.x, lane = tid & 31, w = tid >> 5;
    int m0 = m0b + (w >> 1) * 16;
    int nb = n0b + (w & 1) * 32;
    v8f acc[2] = { zero8(), zero8() };
    for (int k0 = 0; k0 < K; k0 += 32) {
        // speculative prefetch of the next K-chunk (global_prefetch_b8; OOB is
        // silently dropped per ISA, so no guard -> branch-free inner loop)
        __builtin_prefetch(A + (size_t)(m0 + (lane & 15)) * K + k0 + 32, 0, 1);
        __builtin_prefetch(W + (size_t)(nb + (lane & 15)) * K + k0 + 32, 0, 1);
        v16h a = frag_a(A, K, m0, k0);
        acc[0] = wmma16(a, frag_b(W, K, nb, k0), acc[0]);
        acc[1] = wmma16(a, frag_b(W, K, nb + 16, k0), acc[1]);
    }
    int mr = m0 + ((lane >> 4) << 3);
#pragma unroll
    for (int s = 0; s < 2; ++s) {
        int col = nb + s * 16 + (lane & 15);
        float bz = bias ? bias[col] : 0.f;
#pragma unroll
        for (int j = 0; j < 8; ++j) {
            float v = acc[s][j] + bz;
            if (act == 1) v = tanhf(v);
            size_t off = (size_t)(mr + j) * N + col;
            if (outf) outf[off] = v;
            if (outh) outh[off] = (f16)v;
        }
    }
}

// ---------------------------------------------------------------- persistent single-WGP LSTM scan
// gates_t = h_{t-1} @ whh^T (WMMA, h in LDS f16) + xgc (const over t) or xgt (per-t, precomputed)
__global__ __launch_bounds__(1024) void k_lstm_scan(
    const float* __restrict__ xgc,   // (B,4H) f32 or null
    const f16*  __restrict__ xgt,    // (B,T,4H) f16 or null
    const f16*  __restrict__ whh,    // (4H,256) f16
    f16* __restrict__ hs,            // (B,T,256) f16
    float* __restrict__ hsum) {      // (B,256) f32 or null
    extern __shared__ f16 smem[];
    f16* hsh = smem;              // 64 x 256
    f16* gsh = smem + 64 * 256;   // 64 x 1024
    int tid = threadIdx.x, lane = tid & 31, w = tid >> 5;
    for (int i = tid; i < 64 * 256; i += 1024) hsh[i] = (f16)0.f;
    const int b  = tid >> 4;          // cell-state owner: 16 h-elements per thread
    const int j0 = (tid & 15) * 16;
    float c[16], hacc[16];
#pragma unroll
    for (int u = 0; u < 16; ++u) { c[u] = 0.f; hacc[u] = 0.f; }
    const int m0 = (w & 3) * 16;          // 4 M-tiles
    const int n0base = (w >> 2) * 128;    // 8 N-tiles of 16 per wave
    __syncthreads();
    for (int t = 0; t < TLEN; ++t) {
        // --- recurrent GEMM: gates = h @ whh^T (K=256)
        for (int nt = 0; nt < 8; ++nt) {
            int n0 = n0base + nt * 16;
            v8f acc = zero8();
#pragma unroll
            for (int k0 = 0; k0 < 256; k0 += 32) {
                v16h a  = frag_a(hsh, 256, m0, k0);
                v16h bb = frag_b(whh, 256, n0, k0);
                acc = wmma16(a, bb, acc);
            }
            int col = n0 + (lane & 15);
            int mr  = m0 + ((lane >> 4) << 3);
#pragma unroll
            for (int j = 0; j < 8; ++j) gsh[(mr + j) * 1024 + col] = (f16)acc[j];
        }
        __syncthreads();
        // --- elementwise LSTM cell (c in registers)
#pragma unroll
        for (int u = 0; u < 16; ++u) {
            int j = j0 + u;
            float gi = (float)gsh[b * 1024 + j];
            float gf = (float)gsh[b * 1024 + 256 + j];
            float gg = (float)gsh[b * 1024 + 512 + j];
            float go = (float)gsh[b * 1024 + 768 + j];
            if (xgc) {
                const float* p = xgc + b * 1024;
                gi += p[j]; gf += p[256 + j]; gg += p[512 + j]; go += p[768 + j];
            }
            if (xgt) {
                const f16* p = xgt + ((size_t)b * TLEN + t) * 1024;
                gi += (float)p[j]; gf += (float)p[256 + j];
                gg += (float)p[512 + j]; go += (float)p[768 + j];
            }
            float cn = sigm(gf) * c[u] + sigm(gi) * tanhf(gg);
            float hv = sigm(go) * tanhf(cn);
            c[u] = cn; hacc[u] += hv;
            f16 hh = (f16)hv;
            hsh[b * 256 + j] = hh;
            hs[((size_t)b * TLEN + t) * 256 + j] = hh;
        }
        __syncthreads();
    }
    if (hsum)
#pragma unroll
        for (int u = 0; u < 16; ++u) hsum[b * 256 + j0 + u] = hacc[u];
}

// ---------------------------------------------------------------- conv1d as implicit GEMM (time-major activations)
// A-tile (64x32 f16) staged into LDS via async global->LDS when available.
__global__ __launch_bounds__(256) void k_conv_gemm(
    const f16* __restrict__ src0, const f16* __restrict__ src1,
    int C0, int Ctot, int ks, int pad,
    const f16* __restrict__ wpk, const float* __restrict__ scale, const float* __restrict__ shift,
    f16* __restrict__ out, int N) {
    __shared__ f16 Ash[64 * 32];
    int bid = blockIdx.x;
    int bb = bid >> 4;             // T/64 == 16 chunks
    int t0 = (bid & 15) * 64;
    int n0b = blockIdx.y * 64;
    int Kc = Ctot * ks;
    int tid = threadIdx.x, lane = tid & 31, w = tid >> 5;
    int m0 = (w >> 1) * 16;
    int nb = n0b + (w & 1) * 32;
    int C1 = Ctot - C0;
    v8f acc[2] = { zero8(), zero8() };
    for (int kc = 0; kc < Kc; kc += 32) {
        int kk = kc / Ctot;        // kernel tap (chunk never straddles taps: Ctot % 32 == 0)
        int ibase = kc % Ctot;
        __syncthreads();
#if USE_ASYNC_LDS
        {   // one async b128 per lane moves the whole 4KB tile: 256 lanes x 16B
            int r = tid >> 2;
            int seg = (tid & 3) << 3;          // 8-half segment within the row
            int tt = t0 + r + kk - pad;
            int i = ibase + seg;
            f16* dst = &Ash[r * 32 + seg];
            if (tt >= 0 && tt < TLEN) {
                const f16* gsrc = (i < C0)
                    ? (src0 + ((size_t)bb * TLEN + tt) * C0 + i)
                    : (src1 + ((size_t)bb * TLEN + tt) * C1 + (i - C0));
                async_ld_b128(gsrc, dst);
            } else {
                v8h zz;
#pragma unroll
                for (int u = 0; u < 8; ++u) zz[u] = (f16)0.f;
                *(v8h*)dst = zz;
            }
            __builtin_amdgcn_s_wait_asynccnt(0);
        }
#else
        for (int e = tid; e < 2048; e += 256) {
            int r = e >> 5, cc = e & 31;
            int tt = t0 + r + kk - pad;
            int i = ibase + cc;
            f16 v = (f16)0.f;
            if (tt >= 0 && tt < TLEN) {
                if (i < C0) v = src0[((size_t)bb * TLEN + tt) * C0 + i];
                else        v = src1[((size_t)bb * TLEN + tt) * C1 + (i - C0)];
            }
            Ash[r * 32 + cc] = v;
        }
#endif
        __syncthreads();
        v16h a = frag_a(Ash, 32, m0, 0);
        acc[0] = wmma16(a, frag_b(wpk, Kc, nb, kc), acc[0]);
        acc[1] = wmma16(a, frag_b(wpk, Kc, nb + 16, kc), acc[1]);
    }
    int mr = m0 + ((lane >> 4) << 3);
#pragma unroll
    for (int s = 0; s < 2; ++s) {
        int col = nb + s * 16 + (lane & 15);
        float sc = scale[col], sh = shift[col];
#pragma unroll
        for (int j = 0; j < 8; ++j) {
            float v = lrelu(acc[s][j] * sc + sh);
            out[((size_t)bb * TLEN + (t0 + mr + j)) * N + col] = (f16)v;
        }
    }
}

// ---------------------------------------------------------------- small tails
__global__ void k_sinm(const float* sw, const float* sb, float* sinm) {
    int t = blockIdx.x * blockDim.x + threadIdx.x;
    if (t >= TLEN) return;
    const float FR[6] = {0.19f, 0.21f, 0.23f, 0.25f, 0.27f, 0.29f};
    float tt = (float)t / 1023.f;
    float sc[12];
#pragma unroll
    for (int f = 0; f < 6; ++f) {
        float ph = 2.f * PI_F * tt * FR[f] * 1024.f;
        sc[f] = sinf(ph); sc[6 + f] = cosf(ph);
    }
    float m = 0.f;
    for (int o = 0; o < 128; ++o) {
        float a = sb[o];
#pragma unroll
        for (int k = 0; k < 12; ++k) a += sc[k] * sw[o * 12 + k];
        m += a;
    }
    sinm[t] = m * (1.f / 128.f);
}

__global__ __launch_bounds__(128) void k_cp(
    const float* csum, const float* fsum,
    const float* w1, const float* b1, const float* g, const float* be,
    const float* w2, const float* b2, float* cp) {
    __shared__ float red[128];
    __shared__ float sv[128];
    int b = blockIdx.x, o = threadIdx.x;
    const float* wr = w1 + (size_t)o * 512;
    float acc = b1[o];
    for (int k = 0; k < 256; ++k) acc += csum[b * 256 + k] * (1.f / 1024.f) * wr[k];
    for (int k = 0; k < 256; ++k) acc += fsum[b * 256 + k] * (1.f / 1024.f) * wr[256 + k];
    red[o] = acc; __syncthreads();
    for (int s = 64; s > 0; s >>= 1) { if (o < s) red[o] += red[o + s]; __syncthreads(); }
    float mean = red[0] * (1.f / 128.f); __syncthreads();
    float d = acc - mean;
    red[o] = d * d; __syncthreads();
    for (int s = 64; s > 0; s >>= 1) { if (o < s) red[o] += red[o + s]; __syncthreads(); }
    float var = red[0] * (1.f / 128.f);
    sv[o] = lrelu(d * rsqrtf(var + 1e-5f) * g[o] + be[o]);
    __syncthreads();
    if (o < 4) {
        float a2 = b2[o];
        for (int k = 0; k < 128; ++k) a2 += sv[k] * w2[o * 128 + k];
        cp[b * 4 + o] = sigm(a2);
    }
}

__global__ void k_conv3(const f16* x, const float* w, const float* bias, float* base) {
    int idx = blockIdx.x * blockDim.x + threadIdx.x;
    if (idx >= BATCH * TLEN) return;
    int b = idx / TLEN, t = idx % TLEN;
    float acc = bias[0];
#pragma unroll
    for (int k = 0; k < 3; ++k) {
        int tt = t + k - 1;
        if (tt < 0 || tt >= TLEN) continue;
        const f16* xr = x + ((size_t)b * TLEN + tt) * 128;
        for (int i = 0; i < 128; ++i) acc += (float)xr[i] * w[i * 3 + k];
    }
    base[idx] = tanhf(acc);
}

__global__ void k_final(const float* base, const float* osc, const float* sinm,
                        const float* cp, const int* labels,
                        const float* sw, const float* aw, const float* ab,
                        float* out) {
    int idx = blockIdx.x * blockDim.x + threadIdx.x;
    if (idx >= BATCH * TLEN) return;
    int b = idx / TLEN, t = idx % TLEN;
    float f0 = 0.19f + 0.1f * cp[b * 4 + 0];
    float amp = 1.f + 2.f * cp[b * 4 + 1];
    float ph  = 2.f * PI_F * cp[b * 4 + 2];
    float bl  = -0.5f + cp[b * 4 + 3];
    auto ev = [&](int tc) -> float {
        if (tc < 0 || tc >= TLEN) return 0.f;
        float card = amp * sinf(2.f * PI_F * f0 * 1024.f * ((float)tc / 1023.f) + ph) + bl;
        return 0.1f * base[b * TLEN + tc] + 0.1f * osc[b * TLEN + tc] + 0.7f * card + 0.1f * sinm[tc];
    };
    int lab = labels[b];
    float r;
    if (lab == 1)      r = ev(t);
    else if (lab == 2) r = sw[0] * ev(t);
    else if (lab == 3) r = aw[0] * ev(t - 1) + aw[1] * ev(t) + aw[2] * ev(t + 1) + ab[0];
    else               r = 0.f;
    out[idx] = r;
}

// ---------------------------------------------------------------- launch
extern "C" void kernel_launch(void* const* d_in, const int* in_sizes, int n_in,
                              void* d_out, int out_size, void* d_ws, size_t ws_size,
                              hipStream_t stream) {
    const float* z       = (const float*)d_in[0];
    const int*   labels  = (const int*)  d_in[1];
    const float* emb     = (const float*)d_in[2];
    const float* np_w    = (const float*)d_in[3];
    const float* np_b    = (const float*)d_in[4];
    const float* np_g    = (const float*)d_in[5];
    const float* np_be   = (const float*)d_in[6];
    const float* c_wih   = (const float*)d_in[7];
    const float* c_whh   = (const float*)d_in[8];
    const float* c_bih   = (const float*)d_in[9];
    const float* c_bhh   = (const float*)d_in[10];
    const float* f0_wih  = (const float*)d_in[11];
    const float* f0_whh  = (const float*)d_in[12];
    const float* f0_bih  = (const float*)d_in[13];
    const float* f0_bhh  = (const float*)d_in[14];
    const float* f_wih   = (const float*)d_in[15];
    const float* f_whh   = (const float*)d_in[16];
    const float* f_bih   = (const float*)d_in[17];
    const float* f_bhh   = (const float*)d_in[18];
    const float* osc_w1  = (const float*)d_in[19];
    const float* osc_b1  = (const float*)d_in[20];
    const float* osc_g   = (const float*)d_in[21];
    const float* osc_be  = (const float*)d_in[22];
    const float* osc_w2  = (const float*)d_in[23];
    const float* osc_b2  = (const float*)d_in[24];
    const float* cp_w1   = (const float*)d_in[25];
    const float* cp_b1   = (const float*)d_in[26];
    const float* cp_g    = (const float*)d_in[27];
    const float* cp_be   = (const float*)d_in[28];
    const float* cp_w2   = (const float*)d_in[29];
    const float* cp_b2   = (const float*)d_in[30];
    const float* sin_w   = (const float*)d_in[31];
    const float* sin_b   = (const float*)d_in[32];
    const float* conv1_w = (const float*)d_in[33];
    const float* conv1_b = (const float*)d_in[34];
    const float* bn1_g   = (const float*)d_in[35];
    const float* bn1_b   = (const float*)d_in[36];
    const float* conv2_w = (const float*)d_in[37];
    const float* conv2_b = (const float*)d_in[38];
    const float* bn2_g   = (const float*)d_in[39];
    const float* bn2_b   = (const float*)d_in[40];
    const float* conv3_w = (const float*)d_in[41];
    const float* conv3_b = (const float*)d_in[42];
    const float* stress_w= (const float*)d_in[43];
    const float* amuse_w = (const float*)d_in[44];
    const float* amuse_b = (const float*)d_in[45];

    char* ws = (char*)d_ws;
    size_t off = 0;
    auto alloc = [&](size_t bytes) -> char* {
        char* p = ws + off;
        off += (bytes + 255) & ~(size_t)255;
        return p;
    };
    f16* c_wih16  = (f16*)alloc(1024 * 512 * 2);
    f16* f0_wih16 = (f16*)alloc(1024 * 512 * 2);
    f16* c_whh16  = (f16*)alloc(1024 * 256 * 2);
    f16* f0_whh16 = (f16*)alloc(1024 * 256 * 2);
    f16* fwih16[3]; f16* fwhh16[3]; float* fbl[3];
    for (int l = 0; l < 3; ++l) {
        fwih16[l] = (f16*)alloc(1024 * 256 * 2);
        fwhh16[l] = (f16*)alloc(1024 * 256 * 2);
        fbl[l]    = (float*)alloc(1024 * 4);
    }
    f16*   oscw2_16 = (f16*)alloc(1024 * 256 * 2);
    float* cbsum    = (float*)alloc(1024 * 4);
    float* f0bsum   = (float*)alloc(1024 * 4);
    f16*   sig16    = (f16*)alloc(64 * 512 * 2);
    f16*   oscmid16 = (f16*)alloc(64 * 256 * 2);
    float* xg_c     = (float*)alloc(64 * 1024 * 4);
    float* xg_f0    = (float*)alloc(64 * 1024 * 4);
    float* oscbuf   = (float*)alloc(64 * 1024 * 4);
    float* sinm     = (float*)alloc(1024 * 4);
    f16*   coarse_h = (f16*)alloc((size_t)64 * 1024 * 256 * 2);
    f16*   fineA    = (f16*)alloc((size_t)64 * 1024 * 256 * 2);
    f16*   fineB    = (f16*)alloc((size_t)64 * 1024 * 256 * 2);
    f16*   xgt      = (f16*)alloc((size_t)64 * 1024 * 1024 * 2);
    float* csum     = (float*)alloc(64 * 256 * 4);
    float* fsum     = (float*)alloc(64 * 256 * 4);
    float* cpbuf    = (float*)alloc(64 * 4 * 4);
    f16*   w1p      = (f16*)alloc(256 * 2560 * 2);
    float* sc1      = (float*)alloc(256 * 4);
    float* sh1      = (float*)alloc(256 * 4);
    f16*   w2p      = (f16*)alloc(128 * 768 * 2);
    float* sc2      = (float*)alloc(128 * 4);
    float* sh2      = (float*)alloc(128 * 4);
    f16*   out1     = (f16*)alloc((size_t)64 * 1024 * 256 * 2);
    f16*   out2     = (f16*)alloc((size_t)64 * 1024 * 128 * 2);
    float* basebuf  = (float*)alloc(64 * 1024 * 4);
    (void)ws_size; (void)in_sizes; (void)n_in; (void)out_size;

    auto gsz = [](int n) { return (n + 255) / 256; };

    // weight/bias packs
    k_f32_to_f16<<<gsz(1024 * 512), 256, 0, stream>>>(c_wih, c_wih16, 1024 * 512);
    k_f32_to_f16<<<gsz(1024 * 512), 256, 0, stream>>>(f0_wih, f0_wih16, 1024 * 512);
    k_f32_to_f16<<<gsz(1024 * 256), 256, 0, stream>>>(c_whh, c_whh16, 1024 * 256);
    k_f32_to_f16<<<gsz(1024 * 256), 256, 0, stream>>>(f0_whh, f0_whh16, 1024 * 256);
    k_f32_to_f16<<<gsz(1024 * 256), 256, 0, stream>>>(osc_w2, oscw2_16, 1024 * 256);
    for (int l = 0; l < 3; ++l) {
        k_f32_to_f16<<<gsz(1024 * 256), 256, 0, stream>>>(f_wih + (size_t)l * 1024 * 256, fwih16[l], 1024 * 256);
        k_f32_to_f16<<<gsz(1024 * 256), 256, 0, stream>>>(f_whh + (size_t)l * 1024 * 256, fwhh16[l], 1024 * 256);
        k_add2<<<gsz(1024), 256, 0, stream>>>(f_bih + l * 1024, f_bhh + l * 1024, fbl[l], 1024);
    }
    k_add2<<<gsz(1024), 256, 0, stream>>>(c_bih, c_bhh, cbsum, 1024);
    k_add2<<<gsz(1024), 256, 0, stream>>>(f0_bih, f0_bhh, f0bsum, 1024);
    k_pack_conv<<<gsz(256 * 512 * 5), 256, 0, stream>>>(conv1_w, conv1_b, bn1_g, bn1_b, w1p, sc1, sh1, 256, 512, 5);
    k_pack_conv<<<gsz(128 * 256 * 3), 256, 0, stream>>>(conv2_w, conv2_b, bn2_g, bn2_b, w2p, sc2, sh2, 128, 256, 3);

    // heads
    k_np_head<<<64, 256, 0, stream>>>(z, labels, emb, np_w, np_b, np_g, np_be, sig16);
    k_oscmid<<<64, 256, 0, stream>>>(sig16, osc_w1, osc_b1, osc_g, osc_be, oscmid16);
    k_sinm<<<4, 256, 0, stream>>>(sin_w, sin_b, sinm);

    // t-invariant input projections + osc tail (WMMA)
    k_gemm<<<dim3(16, 1), 256, 0, stream>>>(sig16, c_wih16, cbsum, xg_c, nullptr, 1024, 512, 0);
    k_gemm<<<dim3(16, 1), 256, 0, stream>>>(sig16, f0_wih16, f0bsum, xg_f0, nullptr, 1024, 512, 0);
    k_gemm<<<dim3(16, 1), 256, 0, stream>>>(oscmid16, oscw2_16, osc_b2, oscbuf, nullptr, 1024, 256, 1);

    // LSTM scans (persistent single-WGP, 32 waves, h+gates in LDS)
    size_t lds = (size_t)(64 * 256 + 64 * 1024) * sizeof(f16);  // 160 KB
    k_lstm_scan<<<1, 1024, lds, stream>>>(xg_c, nullptr, c_whh16, coarse_h, csum);
    k_lstm_scan<<<1, 1024, lds, stream>>>(xg_f0, nullptr, f0_whh16, fineA, nullptr);
    f16* cur = fineA; f16* nxt = fineB;
    for (int l = 0; l < 3; ++l) {
        // big parallel input projection (65536 x 1024, K=256) -> xgt f16
        k_gemm<<<dim3(16, 1024), 256, 0, stream>>>(cur, fwih16[l], fbl[l], nullptr, xgt, 1024, 256, 0);
        k_lstm_scan<<<1, 1024, lds, stream>>>(nullptr, xgt, fwhh16[l], nxt, (l == 2) ? fsum : nullptr);
        f16* t = cur; cur = nxt; nxt = t;
    } // cur == final fine

    // cardiac params
    k_cp<<<64, 128, 0, stream>>>(csum, fsum, cp_w1, cp_b1, cp_g, cp_be, cp_w2, cp_b2, cpbuf);

    // conv stack as implicit WMMA GEMM
    k_conv_gemm<<<dim3(1024, 4), 256, 0, stream>>>(coarse_h, cur, 256, 512, 5, 2, w1p, sc1, sh1, out1, 256);
    k_conv_gemm<<<dim3(1024, 2), 256, 0, stream>>>(out1, out1, 256, 256, 3, 1, w2p, sc2, sh2, out2, 128);
    k_conv3<<<gsz(64 * 1024), 256, 0, stream>>>(out2, conv3_w, conv3_b, basebuf);

    // combine + label select
    k_final<<<gsz(64 * 1024), 256, 0, stream>>>(basebuf, oscbuf, sinm, cpbuf, labels,
                                                stress_w, amuse_w, amuse_b, (float*)d_out);
}
